// LSTMNetwork_48945447305537
// MI455X (gfx1250) — matrix operand order repaired and checked by reading, not compiled, not run on hardware
//
#include <hip/hip_runtime.h>
#include <hip/hip_bf16.h>

// LSTM on MI455X (gfx1250), wave32 + WMMA bf16 + async global->LDS staging,
// software-pipelined fragment loads.
//
// Phases (all on `stream`, sequential):
//   0) init barrier + h ring buffer; convert x -> bf16; transpose Wx/Wh/Wd -> bf16 [N][K]
//   A) x_proj[M=32768, G=2048] = x @ Wx + b   (block's 8 waves share a WxT slice in LDS,
//      staged via global_load_async_to_lds_b128; A/B fragments double-buffered)
//   B) persistent recurrent kernel: 64 waves, 1024 steps; per-wave Wh/Wd slices live in
//      LDS for the whole sequence (async-staged once); triple-buffered h, one device-scope
//      barrier per step; fused output projection h_t @ Wd + bd -> d_out
//
// ws usage ~172 MB: bar | hbuf(3x32x512 bf16) | x_bf16 | WxT | WhT | WdT | x_proj_bf16

#define LSTM_B   32
#define LSTM_S   1024
#define LSTM_I   512
#define LSTM_H   512
#define LSTM_O   512
#define LSTM_G   2048          // 4*H
#define LSTM_M   (LSTM_B * LSTM_S)
#define NWG      64            // persistent workgroups (1 wave each)

// LDS row pitch: 512 bf16 + 8 pad -> 1040 B; 16-lane b128 fragment reads hit all
// 64 banks exactly twice (minimum) -> conflict-free.
#define LDS_PITCH_E 520
#define LDS_PITCH_B (LDS_PITCH_E * 2)

typedef __attribute__((ext_vector_type(16))) __bf16 v16bf;
typedef __attribute__((ext_vector_type(8)))  float  v8f;

union Frag16 { v16bf v; uint4 q[2]; };

// A-fragment (16x32 bf16, MxK). lane<16 -> row m, K = {0..7, 16..23};
// lane>=16 -> row m, K = {8..15, 24..31}. rowbase = A + m*K + k0 + half*8.
__device__ __forceinline__ v16bf load_a_frag(const __bf16* rowbase) {
    Frag16 f;
    f.q[0] = *reinterpret_cast<const uint4*>(rowbase);        // k .. k+7
    f.q[1] = *reinterpret_cast<const uint4*>(rowbase + 16);   // k+16 .. k+23
    return f.v;
}

// B-fragment (32x16 bf16, KxN) from LDS: lane holds column n; lanes 0-15 K={0..7,16..23},
// lanes 16-31 K={8..15,24..31}. rowbase already includes k0+half*8.
__device__ __forceinline__ v16bf load_b_frag_lds(const unsigned char* rowbase) {
    Frag16 f;
    const uint4* p = reinterpret_cast<const uint4*>(rowbase);
    f.q[0] = p[0];
    f.q[1] = p[2];
    return f.v;
}

// CDNA5 async copy: 16B global -> LDS per lane, tracked by ASYNCcnt.
__device__ __forceinline__ void async_copy_b128(unsigned lds_off, const void* gptr) {
    unsigned long long ga = (unsigned long long)(size_t)gptr;
    asm volatile("global_load_async_to_lds_b128 %0, %1, off"
                 :: "v"(lds_off), "v"(ga) : "memory");
}
__device__ __forceinline__ void wait_async0() {
    asm volatile("s_wait_asynccnt 0x0" ::: "memory");
}

__device__ __forceinline__ float sigmoidf(float x) {
    return 1.0f / (1.0f + __expf(-x));
}

// ---------------------------------------------------------------- init / convert

__global__ void init_kernel(unsigned* __restrict__ bar, __bf16* __restrict__ hbuf, int n) {
    int i = blockIdx.x * blockDim.x + threadIdx.x;
    if (i < 2) bar[i] = 0u;
    if (i < n) hbuf[i] = (__bf16)0.0f;
}

__global__ void f32_to_bf16_kernel(const float* __restrict__ in, __bf16* __restrict__ out, int n) {
    int i = blockIdx.x * blockDim.x + threadIdx.x;
    if (i < n) out[i] = (__bf16)in[i];
}

// in: [K][N] f32 row-major  ->  out: [N][K] bf16 (K-major per output row)
__global__ void transpose_to_bf16_kernel(const float* __restrict__ in, __bf16* __restrict__ out,
                                         int K, int N) {
    int i = blockIdx.x * blockDim.x + threadIdx.x;
    if (i < K * N) {
        int k = i / N;
        int n = i - k * N;
        out[n * K + k] = (__bf16)in[i];
    }
}

// ---------------------------------------------------------------- phase A: x @ Wx + b

// 8 waves/block. All waves of a block share one 64-column WxT slice (ng) staged in
// LDS; wave wid handles row tile mt = mtBase + wid. Grid: 256 mt-groups x 32 ng.
__global__ void xproj_kernel(const __bf16* __restrict__ xb,   // [M][I] bf16
                             const __bf16* __restrict__ WxT,  // [G][I] bf16
                             const float*  __restrict__ bias, // [G]
                             __bf16* __restrict__ P)          // [M][G] bf16
{
    extern __shared__ __align__(16) unsigned char smem[];

    const int tid    = threadIdx.x;
    const int lane   = tid & 31;
    const int wid    = tid >> 5;
    const int mt     = (blockIdx.x & 255) * 8 + wid;  // 0..2047
    const int ng     = blockIdx.x >> 8;               // 0..31
    const int lane15 = lane & 15;
    const int half   = lane >> 4;

    // ---- stage WxT rows [ng*64, ng*64+64) into LDS (64 rows x 512 bf16), async
    const unsigned smem_base = (unsigned)(size_t)smem;
    for (int ci = tid; ci < 64 * 64; ci += 256) {          // 16B chunks
        int r  = ci >> 6;       // local row 0..63
        int ck = ci & 63;       // 16B chunk within row
        async_copy_b128(smem_base + (unsigned)(r * LDS_PITCH_B + ck * 16),
                        WxT + (ng * 64 + r) * LSTM_I + ck * 8);
    }
    wait_async0();
    __syncthreads();

    v8f acc[4];
    #pragma unroll
    for (int t = 0; t < 4; ++t) {
        float bb = bias[ng * 64 + t * 16 + lane15];
        #pragma unroll
        for (int r = 0; r < 8; ++r) acc[t][r] = bb;
    }

    const __bf16* arow = xb + (mt * 16 + lane15) * LSTM_I + half * 8;
    const unsigned char* bbase = smem + (size_t)(half * 16);   // (half*8 elems)*2B

    // software pipeline: fragments for k+32 load while WMMAs for k execute
    v16bf a_cur = load_a_frag(arow);
    v16bf b_cur[4];
    #pragma unroll
    for (int t = 0; t < 4; ++t)
        b_cur[t] = load_b_frag_lds(bbase + (t * 16 + lane15) * LDS_PITCH_B);

    for (int k0 = 0; k0 < LSTM_I; k0 += 32) {
        const int kn = (k0 + 32 < LSTM_I) ? (k0 + 32) : 0;   // wrap: keeps pipe branch-free
        v16bf a_next = load_a_frag(arow + kn);
        v16bf b_next[4];
        #pragma unroll
        for (int t = 0; t < 4; ++t)
            b_next[t] = load_b_frag_lds(bbase + (t * 16 + lane15) * LDS_PITCH_B + kn * 2);
        #pragma unroll
        for (int t = 0; t < 4; ++t)
            acc[t] = __builtin_amdgcn_wmma_f32_16x16x32_bf16(
                false, a_cur, false, b_cur[t], (short)0, acc[t], false, false);
        a_cur = a_next;
        #pragma unroll
        for (int t = 0; t < 4; ++t) b_cur[t] = b_next[t];
    }

    #pragma unroll
    for (int t = 0; t < 4; ++t)
        #pragma unroll
        for (int r = 0; r < 8; ++r)
            P[(mt * 16 + half * 8 + r) * LSTM_G + ng * 64 + t * 16 + lane15] = (__bf16)acc[t][r];
}

// ---------------------------------------------------------------- phase B: recurrence

__device__ __forceinline__ void grid_barrier(unsigned* count, unsigned* gen, unsigned step) {
    __threadfence();
    if (threadIdx.x == 0) {
        unsigned prev = __hip_atomic_fetch_add(count, 1u, __ATOMIC_ACQ_REL, __HIP_MEMORY_SCOPE_AGENT);
        if (prev == NWG - 1u) {
            __hip_atomic_store(count, 0u, __ATOMIC_RELAXED, __HIP_MEMORY_SCOPE_AGENT);
            __hip_atomic_store(gen, step + 1u, __ATOMIC_RELEASE, __HIP_MEMORY_SCOPE_AGENT);
        } else {
            while (__hip_atomic_load(gen, __ATOMIC_ACQUIRE, __HIP_MEMORY_SCOPE_AGENT) < step + 1u) {
                __builtin_amdgcn_s_sleep(8);
            }
        }
    }
    __syncthreads();
}

// 64 blocks x 32 threads (1 wave/block). Block (mt, hct) owns gate columns
// [hct*16,+16) of all four gates for batch rows [mt*16,+16). Its WhT slice
// (4x16 rows x 512) and WdT slice (16 rows x 512) are staged into LDS once and
// reused for all 1024 steps. Cell state c stays in registers; h triple-buffered.
__global__ void lstm_recurrent_kernel(const __bf16* __restrict__ P,    // [M][G] bf16 (x_proj)
                                      const __bf16* __restrict__ WhT,  // [G][H] bf16
                                      const __bf16* __restrict__ WdT,  // [O][H] bf16
                                      const float*  __restrict__ bias, // [G]
                                      const float*  __restrict__ bd,   // [O]
                                      __bf16* __restrict__ hbuf,       // [3][B][H] bf16
                                      float* __restrict__ out,         // [B][S][O] f32
                                      unsigned* __restrict__ bar)      // {count, gen}
{
    extern __shared__ __align__(16) unsigned char smem[];

    const int lane   = threadIdx.x & 31;
    const int mt     = blockIdx.x >> 5;   // 0..1
    const int hct    = blockIdx.x & 31;   // 0..31
    const int lane15 = lane & 15;
    const int half   = lane >> 4;
    const int col    = hct * 16 + lane15; // h / out column owned by this lane

    // ---- stage weights into LDS once (async): rows 0..63 = Wh gate slices,
    //      rows 64..79 = Wd slice; each row 512 bf16, pitch LDS_PITCH_B.
    const unsigned smem_base = (unsigned)(size_t)smem;
    for (int ci = lane; ci < 80 * 64; ci += 32) {          // 16B chunks
        int r  = ci >> 6;       // local row 0..79
        int ck = ci & 63;
        const __bf16* g;
        if (r < 64) g = WhT + ((r >> 4) * LSTM_H + hct * 16 + (r & 15)) * LSTM_H + ck * 8;
        else        g = WdT + (hct * 16 + (r - 64)) * LSTM_H + ck * 8;
        async_copy_b128(smem_base + (unsigned)(r * LDS_PITCH_B + ck * 16), g);
    }
    wait_async0();   // single wave per block: no further barrier needed

    float bz[4];
    #pragma unroll
    for (int g = 0; g < 4; ++g) bz[g] = bias[g * LSTM_H + col];
    const float bo = bd[col];

    float c[8];
    #pragma unroll
    for (int r = 0; r < 8; ++r) c[r] = 0.0f;

    const unsigned char* bbase = smem + (size_t)(half * 16);  // + (half*8 elems)*2B

    for (int t = 0; t < LSTM_S; ++t) {
        const __bf16* hprev = hbuf + ((t + 2) % 3) * (LSTM_B * LSTM_H);
        __bf16*       hcur  = hbuf + (t % 3)       * (LSTM_B * LSTM_H);

        // x_proj gate terms: independent of the GEMM -> issue loads first so their
        // L2 latency hides under the 64 WMMAs below.
        float px[4][8];
        #pragma unroll
        for (int r = 0; r < 8; ++r) {
            const int     brow = mt * 16 + half * 8 + r;                 // batch index
            const __bf16* prow = P + (size_t)(brow * LSTM_S + t) * LSTM_G;
            #pragma unroll
            for (int g = 0; g < 4; ++g) px[g][r] = (float)prow[g * LSTM_H + col];
        }

        // z = h_{t-1} @ Wh (+bias): 4 gate tiles share one A fragment; B from LDS;
        // fragments double-buffered across k.
        v8f acc[4];
        #pragma unroll
        for (int g = 0; g < 4; ++g)
            #pragma unroll
            for (int r = 0; r < 8; ++r) acc[g][r] = bz[g];

        const __bf16* harow = hprev + (mt * 16 + lane15) * LSTM_H + half * 8;
        v16bf a_cur = load_a_frag(harow);
        v16bf b_cur[4];
        #pragma unroll
        for (int g = 0; g < 4; ++g)
            b_cur[g] = load_b_frag_lds(bbase + (g * 16 + lane15) * LDS_PITCH_B);

        for (int k0 = 0; k0 < LSTM_H; k0 += 32) {
            const int kn = (k0 + 32 < LSTM_H) ? (k0 + 32) : 0;
            v16bf a_next = load_a_frag(harow + kn);
            v16bf b_next[4];
            #pragma unroll
            for (int g = 0; g < 4; ++g)
                b_next[g] = load_b_frag_lds(bbase + (g * 16 + lane15) * LDS_PITCH_B + kn * 2);
            #pragma unroll
            for (int g = 0; g < 4; ++g)
                acc[g] = __builtin_amdgcn_wmma_f32_16x16x32_bf16(
                    false, a_cur, false, b_cur[g], (short)0, acc[g], false, false);
            a_cur = a_next;
            #pragma unroll
            for (int g = 0; g < 4; ++g) b_cur[g] = b_next[g];
        }

        // gates + state update (flax order i, f, g, o); write h_t
        #pragma unroll
        for (int r = 0; r < 8; ++r) {
            const int brow = mt * 16 + half * 8 + r;
            float zi = acc[0][r] + px[0][r];
            float zf = acc[1][r] + px[1][r];
            float zg = acc[2][r] + px[2][r];
            float zo = acc[3][r] + px[3][r];
            float cn = sigmoidf(zf) * c[r] + sigmoidf(zi) * tanhf(zg);
            c[r] = cn;
            float hn = sigmoidf(zo) * tanhf(cn);
            hcur[brow * LSTM_H + col] = (__bf16)hn;
        }

        grid_barrier(bar, bar + 1, (unsigned)t);   // h_t now globally visible

        // fused output projection: out[:, t, :] tile = h_t @ Wd + bd (B from LDS),
        // fragments double-buffered across k.
        v8f ao;
        #pragma unroll
        for (int r = 0; r < 8; ++r) ao[r] = bo;
        const __bf16* oarow = hcur + (mt * 16 + lane15) * LSTM_H + half * 8;
        const unsigned char* obrow = bbase + (64 + lane15) * LDS_PITCH_B;
        v16bf oa_cur = load_a_frag(oarow);
        v16bf ob_cur = load_b_frag_lds(obrow);
        for (int k0 = 0; k0 < LSTM_H; k0 += 32) {
            const int kn = (k0 + 32 < LSTM_H) ? (k0 + 32) : 0;
            v16bf oa_next = load_a_frag(oarow + kn);
            v16bf ob_next = load_b_frag_lds(obrow + kn * 2);
            ao = __builtin_amdgcn_wmma_f32_16x16x32_bf16(
                false, oa_cur, false, ob_cur, (short)0, ao, false, false);
            oa_cur = oa_next;
            ob_cur = ob_next;
        }
        #pragma unroll
        for (int r = 0; r < 8; ++r) {
            const int brow = mt * 16 + half * 8 + r;
            out[(size_t)(brow * LSTM_S + t) * LSTM_O + col] = ao[r];
        }
    }
}

// ---------------------------------------------------------------- launcher

extern "C" void kernel_launch(void* const* d_in, const int* in_sizes, int n_in,
                              void* d_out, int out_size, void* d_ws, size_t ws_size,
                              hipStream_t stream) {
    (void)in_sizes; (void)n_in; (void)out_size; (void)ws_size;

    const float* x    = (const float*)d_in[0];   // [B,S,I]
    const float* Wx   = (const float*)d_in[1];   // [I,4H]
    const float* Wh   = (const float*)d_in[2];   // [H,4H]
    const float* b    = (const float*)d_in[3];   // [4H]
    const float* Wd   = (const float*)d_in[4];   // [H,O]
    const float* bd   = (const float*)d_in[5];   // [O]
    float*       outp = (float*)d_out;           // [B,S,O]

    char* w = (char*)d_ws;
    size_t off = 0;
    auto take = [&](size_t bytes) -> void* {
        off = (off + 255) & ~(size_t)255;
        void* p = w + off;
        off += bytes;
        return p;
    };

    unsigned* bar  = (unsigned*)take(256);
    __bf16*   hbuf = (__bf16*)take(sizeof(__bf16) * 3 * LSTM_B * LSTM_H);
    __bf16*   xb   = (__bf16*)take(sizeof(__bf16) * (size_t)LSTM_M * LSTM_I);
    __bf16*   WxT  = (__bf16*)take(sizeof(__bf16) * (size_t)LSTM_G * LSTM_I);
    __bf16*   WhT  = (__bf16*)take(sizeof(__bf16) * (size_t)LSTM_G * LSTM_H);
    __bf16*   WdT  = (__bf16*)take(sizeof(__bf16) * (size_t)LSTM_O * LSTM_H);
    __bf16*   P    = (__bf16*)take(sizeof(__bf16) * (size_t)LSTM_M * LSTM_G);

    // phase 0: init + precision conversion / weight transposition
    {
        int n = 3 * LSTM_B * LSTM_H;
        init_kernel<<<(n + 255) / 256, 256, 0, stream>>>(bar, hbuf, n);
    }
    {
        int n = LSTM_M * LSTM_I;
        f32_to_bf16_kernel<<<(n + 255) / 256, 256, 0, stream>>>(x, xb, n);
    }
    transpose_to_bf16_kernel<<<(LSTM_I * LSTM_G + 255) / 256, 256, 0, stream>>>(Wx, WxT, LSTM_I, LSTM_G);
    transpose_to_bf16_kernel<<<(LSTM_H * LSTM_G + 255) / 256, 256, 0, stream>>>(Wh, WhT, LSTM_H, LSTM_G);
    transpose_to_bf16_kernel<<<(LSTM_H * LSTM_O + 255) / 256, 256, 0, stream>>>(Wd, WdT, LSTM_H, LSTM_O);

    // phase A: x_proj = x @ Wx + b  (8 waves/block sharing a 64-col WxT LDS slice)
    {
        const int blocks = 256 * 32;                 // mt-groups x ng
        const int smem   = 64 * LDS_PITCH_B;         // 66,560 B
        xproj_kernel<<<blocks, 256, smem, stream>>>(xb, WxT, b, P);
    }

    // phase B: persistent recurrence + fused output projection
    {
        const int smem = 80 * LDS_PITCH_B;           // 83,200 B
        lstm_recurrent_kernel<<<NWG, 32, smem, stream>>>(P, WhT, WdT, b, bd, hbuf, outp, bar);
    }
}
